// MixtralSparseMoeBlock_76751065579770
// MI455X (gfx1250) — compile-verified, matrix operand order checked by reading, not run on hardware
//
#include <hip/hip_runtime.h>
#include <hip/hip_bf16.h>

#define T_TOK 8192
#define H_DIM 1024
#define I_DIM 3584
#define N_EXP 8
#define TOP_K 2

#define TILE_M 16          // tokens per block
#define I_CHUNK 256        // intermediate columns per fused phase (8 waves * 32)
#define TILES_PER_E (T_TOK / TILE_M)   // 512

#define LDX 1040           // padded bf16 stride for x tile in LDS (2080B, 16B aligned)
#define LDH 264            // padded bf16 stride for h tile in LDS (528B, 16B aligned)

typedef __attribute__((ext_vector_type(16))) __bf16 v16bf;
typedef __attribute__((ext_vector_type(8)))  float  v8f;

// ---------------------------------------------------------------------------
// Router: logits -> fp32 softmax -> top-2 (no renormalization), scatter to
// per-expert token lists.  One wave (32 lanes) per token, float4 loads.
// ---------------------------------------------------------------------------
__global__ void router_kernel(const float* __restrict__ x,
                              const float* __restrict__ gate_w,
                              const float* __restrict__ gate_b,
                              int* __restrict__ counts,
                              int* __restrict__ tok_lists,
                              float* __restrict__ wt_lists) {
    const int lane = threadIdx.x & 31;
    const int t    = blockIdx.x * (blockDim.x >> 5) + (threadIdx.x >> 5);
    if (t >= T_TOK) return;

    const float4* xr = (const float4*)(x + (size_t)t * H_DIM);

    float logits[N_EXP];
#pragma unroll
    for (int e = 0; e < N_EXP; ++e) {
        const float4* gr = (const float4*)(gate_w + (size_t)e * H_DIM);
        float s = 0.f;
        for (int k4 = lane; k4 < H_DIM / 4; k4 += 32) {
            float4 a = xr[k4];
            float4 b = gr[k4];
            s += a.x * b.x + a.y * b.y + a.z * b.z + a.w * b.w;
        }
        // wave32 xor-reduction
#pragma unroll
        for (int off = 16; off > 0; off >>= 1)
            s += __shfl_xor(s, off, 32);
        logits[e] = s + gate_b[e];
    }

    if (lane == 0) {
        // fp32 softmax over 8 experts
        float mx = logits[0];
#pragma unroll
        for (int e = 1; e < N_EXP; ++e) mx = fmaxf(mx, logits[e]);
        float den = 0.f;
        float p[N_EXP];
#pragma unroll
        for (int e = 0; e < N_EXP; ++e) { p[e] = __expf(logits[e] - mx); den += p[e]; }
        float inv = 1.f / den;
#pragma unroll
        for (int e = 0; e < N_EXP; ++e) p[e] *= inv;

        // top-2 (no renormalization, matching reference)
        int   i0 = 0; float p0 = p[0];
#pragma unroll
        for (int e = 1; e < N_EXP; ++e) if (p[e] > p0) { p0 = p[e]; i0 = e; }
        int   i1 = -1; float p1 = -1.f;
#pragma unroll
        for (int e = 0; e < N_EXP; ++e)
            if (e != i0 && p[e] > p1) { p1 = p[e]; i1 = e; }

        int s0 = atomicAdd(&counts[i0], 1);
        tok_lists[i0 * T_TOK + s0] = t;
        wt_lists [i0 * T_TOK + s0] = p0;
        int s1 = atomicAdd(&counts[i1], 1);
        tok_lists[i1 * T_TOK + s1] = t;
        wt_lists [i1 * T_TOK + s1] = p1;
    }
}

// ---------------------------------------------------------------------------
// WMMA fragment loaders (bf16, 16x16x32 shapes).
// A 16x32 layout: lanes 0-15 hold M=lane, K = {kc..kc+7, kc+16..kc+23};
//                 lanes 16-31 hold M=lane-16, K = {kc+8..kc+15, kc+24..kc+31}.
// B 32x16 uses the symmetric layout with lane -> column index.
// All K-groups are 8 contiguous elements at 16B-aligned addresses, so we use
// ds_load_b128 (LDS) and global_load_b128 (weights) wide accesses.
// ---------------------------------------------------------------------------
__device__ __forceinline__ v16bf load_frag_lds(const __bf16* __restrict__ base,
                                               int ld, int kc, int lane) {
    const int m  = lane & 15;
    const int kb = kc + ((lane & 16) ? 8 : 0);
    const __bf16* p = base + m * ld + kb;
    union { uint4 u[2]; v16bf f; } t;
    t.u[0] = *(const uint4*)(p);        // K = kb .. kb+7
    t.u[1] = *(const uint4*)(p + 16);   // K = kb+16 .. kb+23
    return t.f;
}

__device__ __forceinline__ v16bf load_frag_w(const float* __restrict__ wbase,
                                             int ld, int colbase, int kc, int lane) {
    const int n  = colbase + (lane & 15);
    const int kb = kc + ((lane & 16) ? 8 : 0);
    const float* p = wbase + (size_t)n * ld + kb;
    float4 q0 = ((const float4*)p)[0];          // K = kb .. kb+3
    float4 q1 = ((const float4*)p)[1];          // K = kb+4 .. kb+7
    float4 q2 = ((const float4*)(p + 16))[0];   // K = kb+16 .. kb+19
    float4 q3 = ((const float4*)(p + 16))[1];   // K = kb+20 .. kb+23
    v16bf f;
    f[0]  = (__bf16)q0.x; f[1]  = (__bf16)q0.y; f[2]  = (__bf16)q0.z; f[3]  = (__bf16)q0.w;
    f[4]  = (__bf16)q1.x; f[5]  = (__bf16)q1.y; f[6]  = (__bf16)q1.z; f[7]  = (__bf16)q1.w;
    f[8]  = (__bf16)q2.x; f[9]  = (__bf16)q2.y; f[10] = (__bf16)q2.z; f[11] = (__bf16)q2.w;
    f[12] = (__bf16)q3.x; f[13] = (__bf16)q3.y; f[14] = (__bf16)q3.z; f[15] = (__bf16)q3.w;
    return f;
}

__device__ __forceinline__ v8f wmma_bf16(v16bf a, v16bf b, v8f c) {
    return __builtin_amdgcn_wmma_f32_16x16x32_bf16(
        /*neg_a=*/false, a, /*neg_b=*/false, b,
        /*c_mod=*/(short)0, c, /*reuse_a=*/false, /*reuse_b=*/false);
}

// fast silu: v * rcp(1 + exp(-v))  (v_exp_f32 + v_rcp_f32, no IEEE div sequence)
__device__ __forceinline__ float silu_f(float v) {
    return v * __builtin_amdgcn_rcpf(1.f + __expf(-v));
}

// ---------------------------------------------------------------------------
// Fused expert SwiGLU GEMM.  Block = 16 gathered tokens of one expert.
// 8 waves.  Per 256-wide chunk of I:
//   phase 1: wave w computes h[16 x 32] = silu(x w1^T) * (x w3^T) for its
//            32 intermediate cols (WMMA bf16, K over H=1024) -> bf16 in LDS
//   phase 2: every wave accumulates its 16x128 out slice += h (16x256) w2^T
// Finally scale rows by routing weight and atomicAdd into out.
// ---------------------------------------------------------------------------
__global__ void __launch_bounds__(256, 2)
moe_expert_kernel(const float* __restrict__ x,
                  const float* __restrict__ w1,
                  const float* __restrict__ w2,
                  const float* __restrict__ w3,
                  const int*  __restrict__ counts,
                  const int*  __restrict__ tok_lists,
                  const float* __restrict__ wt_lists,
                  float* __restrict__ out) {
    __shared__ __bf16 xs[TILE_M * LDX];        // gathered token rows (bf16)
    __shared__ __bf16 hs[TILE_M * LDH];        // SwiGLU intermediate tile
    __shared__ int    s_tok[TILE_M];
    __shared__ float  s_wt[TILE_M];

    const int e    = blockIdx.x / TILES_PER_E;
    const int tile = blockIdx.x % TILES_PER_E;
    const int cnt  = counts[e];
    if (tile * TILE_M >= cnt) return;          // uniform early exit

    const int tid  = threadIdx.x;
    const int lane = tid & 31;
    const int wv   = tid >> 5;                 // wave 0..7

    // gather token ids / routing weights for this tile (pad rows -> weight 0)
    if (tid < TILE_M) {
        int slot = tile * TILE_M + tid;
        if (slot < cnt) {
            s_tok[tid] = tok_lists[e * T_TOK + slot];
            s_wt[tid]  = wt_lists [e * T_TOK + slot];
        } else {
            s_tok[tid] = 0;
            s_wt[tid]  = 0.f;
        }
    }
    __syncthreads();

    // stage 16 x 1024 activations into LDS as bf16 (float4 in, packed bf16 out)
    for (int idx = tid; idx < TILE_M * (H_DIM / 4); idx += 256) {
        int m = idx >> 8;                 // / (H_DIM/4)
        int k = (idx & 255) * 4;
        float4 q = *(const float4*)(x + (size_t)s_tok[m] * H_DIM + k);
        union { __bf16 h[4]; uint2 u; } t;
        t.h[0] = (__bf16)q.x; t.h[1] = (__bf16)q.y;
        t.h[2] = (__bf16)q.z; t.h[3] = (__bf16)q.w;
        *(uint2*)&xs[m * LDX + k] = t.u;
    }
    __syncthreads();

    const float* w1e = w1 + (size_t)e * I_DIM * H_DIM;
    const float* w3e = w3 + (size_t)e * I_DIM * H_DIM;
    const float* w2e = w2 + (size_t)e * H_DIM * I_DIM;

    // persistent output accumulators: 16 x 128 per wave (8 tiles of 16x16)
    v8f accO[8];
#pragma unroll
    for (int c = 0; c < 8; ++c) accO[c] = (v8f){};

    const int mrow = (lane & 16) ? 8 : 0;

    for (int ib = 0; ib < I_DIM; ib += I_CHUNK) {
        // ---- phase 1: this wave's 32 intermediate columns ----
        const int icol0 = ib + wv * 32;
        v8f a1a = (v8f){}, a1b = (v8f){}, a3a = (v8f){}, a3b = (v8f){};
        for (int kc = 0; kc < H_DIM; kc += 32) {
            v16bf af  = load_frag_lds(xs, LDX, kc, lane);
            v16bf b1a = load_frag_w(w1e, H_DIM, icol0,      kc, lane);
            v16bf b1b = load_frag_w(w1e, H_DIM, icol0 + 16, kc, lane);
            v16bf b3a = load_frag_w(w3e, H_DIM, icol0,      kc, lane);
            v16bf b3b = load_frag_w(w3e, H_DIM, icol0 + 16, kc, lane);
            // prefetch next K slab of the weight stream
            __builtin_prefetch(w1e + (size_t)(icol0 + (lane & 15)) * H_DIM + kc + 32, 0, 1);
            __builtin_prefetch(w3e + (size_t)(icol0 + (lane & 15)) * H_DIM + kc + 32, 0, 1);
            a1a = wmma_bf16(af, b1a, a1a);
            a1b = wmma_bf16(af, b1b, a1b);
            a3a = wmma_bf16(af, b3a, a3a);
            a3b = wmma_bf16(af, b3b, a3b);
        }

        // silu(gate) * up -> bf16 h tile in LDS
        {
            const int n_loc = wv * 32 + (lane & 15);
#pragma unroll
            for (int v = 0; v < 8; ++v) {
                int m = v + mrow;
                hs[m * LDH + n_loc]      = (__bf16)(silu_f(a1a[v]) * a3a[v]);
                hs[m * LDH + n_loc + 16] = (__bf16)(silu_f(a1b[v]) * a3b[v]);
            }
        }
        __syncthreads();

        // ---- phase 2: out[16 x 128 (this wave)] += h[16 x 256] * w2^T ----
        const int ncol0 = wv * 128;
#pragma unroll 1
        for (int ks = 0; ks < I_CHUNK; ks += 32) {
            v16bf af = load_frag_lds(hs, LDH, ks, lane);
#pragma unroll
            for (int c = 0; c < 8; ++c) {
                v16bf bf = load_frag_w(w2e, I_DIM, ncol0 + c * 16, ib + ks, lane);
                accO[c] = wmma_bf16(af, bf, accO[c]);
            }
        }
        __syncthreads();   // protect hs before next chunk overwrites it
    }

    // ---- combine: scale by routing weight, atomic add into output ----
#pragma unroll
    for (int c = 0; c < 8; ++c) {
        int n = wv * 128 + c * 16 + (lane & 15);
#pragma unroll
        for (int v = 0; v < 8; ++v) {
            int m = v + mrow;
            float val = accO[c][v] * s_wt[m];
            atomicAdd(&out[(size_t)s_tok[m] * H_DIM + n], val);
        }
    }
}

// ---------------------------------------------------------------------------
// Host launcher
// ---------------------------------------------------------------------------
extern "C" void kernel_launch(void* const* d_in, const int* in_sizes, int n_in,
                              void* d_out, int out_size, void* d_ws, size_t ws_size,
                              hipStream_t stream) {
    const float* x      = (const float*)d_in[0];   // [T,H]
    const float* gate_w = (const float*)d_in[1];   // [E,H]
    const float* gate_b = (const float*)d_in[2];   // [E]
    const float* w1     = (const float*)d_in[3];   // [E,I,H]
    const float* w2     = (const float*)d_in[4];   // [E,H,I]
    const float* w3     = (const float*)d_in[5];   // [E,I,H]
    float* out          = (float*)d_out;           // [T,H]

    char* ws = (char*)d_ws;
    int*   counts    = (int*)ws;                                   // 256 B
    int*   tok_lists = (int*)(ws + 256);                           // E*T ints
    float* wt_lists  = (float*)(ws + 256 + (size_t)N_EXP * T_TOK * sizeof(int));

    hipMemsetAsync(counts, 0, 256, stream);
    hipMemsetAsync(out, 0, (size_t)T_TOK * H_DIM * sizeof(float), stream);

    router_kernel<<<T_TOK / 8, 256, 0, stream>>>(x, gate_w, gate_b,
                                                 counts, tok_lists, wt_lists);

    moe_expert_kernel<<<N_EXP * TILES_PER_E, 256, 0, stream>>>(
        x, w1, w2, w3, counts, tok_lists, wt_lists, out);
}
